// IntensityCrossAttention_12326556140273
// MI455X (gfx1250) — matrix operand order repaired and checked by reading, not compiled, not run on hardware
//
#include <hip/hip_runtime.h>
#include <hip/hip_bf16.h>
#include <stdint.h>

typedef __attribute__((ext_vector_type(16))) __bf16 v16bf;
typedef __attribute__((ext_vector_type(8)))  float  v8f;

#define B_      2
#define C_      48
#define HW_     65536
#define QKVC_   240
#define HEADS_  8
#define CC_     6
#define FAC_    8
#define HWF_    8192
#define ROWS_   48

// ---------------------------------------------------------------------------
// WMMA helpers (CDNA5 bf16 16x16x32, wave32)
// ---------------------------------------------------------------------------
__device__ __forceinline__ v8f wmma_bf16(v16bf a, v16bf b, v8f c) {
  return __builtin_amdgcn_wmma_f32_16x16x32_bf16(false, a, false, b, (short)0, c,
                                                 false, false);
}

// A matrix 16xK, row-major in LDS with leading dim `ld`.
__device__ __forceinline__ v16bf frag_a_lds(const __bf16* base, int ld, int lane,
                                            int kofs) {
  int m  = lane & 15;
  int kb = ((lane >> 4) << 4) + kofs;
  const __bf16* row = base + m * ld;
  v16bf f;
#pragma unroll
  for (int v = 0; v < 8; ++v) {
    int k = kb + 2 * v;
    f[2 * v]     = row[k];
    f[2 * v + 1] = row[k + 1];
  }
  return f;
}

// B matrix KxN staged transposed in LDS as [n][k] with leading dim `ld`.
__device__ __forceinline__ v16bf frag_b_lds(const __bf16* base, int ld, int lane,
                                            int kofs, int nsub) {
  int n  = (lane & 15) + nsub;
  int kb = ((lane >> 4) << 4) + kofs;
  const __bf16* p = base + n * ld + kb;
  v16bf f;
#pragma unroll
  for (int v = 0; v < 8; ++v) {
    f[2 * v]     = p[2 * v];
    f[2 * v + 1] = p[2 * v + 1];
  }
  return f;
}

// CDNA5 async global->LDS copy (ASYNCcnt tracked). lds_off is the wave-relative
// LDS byte offset (low 32 bits of the generic pointer per the aperture scheme).
__device__ __forceinline__ void async_g2l_b128(uint32_t lds_off, const void* g) {
  asm volatile("global_load_async_to_lds_b128 %0, %1, off"
               :: "v"(lds_off), "v"(g) : "memory");
}
__device__ __forceinline__ void wait_async0() {
  asm volatile("s_wait_asynccnt 0" ::: "memory");
}
__device__ __forceinline__ uint32_t lds_off32(const void* p) {
  return (uint32_t)(uintptr_t)p;
}

// ---------------------------------------------------------------------------
// 256-element in-LDS bitonic sort (h-columns / w-rows of first 24 channels)
// ---------------------------------------------------------------------------
__global__ void sort256_kernel(float* __restrict__ X, int* __restrict__ IDX,
                               int mode) {
  __shared__ float sv[256];
  __shared__ int   si[256];
  int line = blockIdx.x;
  int p    = line & 255;
  int ch   = (line >> 8) % 24;
  int b    = line / (256 * 24);
  int stride = mode ? 1 : 256;
  int ofs    = mode ? p * 256 : p;
  size_t xbase = ((size_t)(b * 48 + ch) << 16) + ofs;
  size_t ibase = ((size_t)(b * 24 + ch) << 16) + ofs;
  int t = threadIdx.x;
  sv[t] = X[xbase + (size_t)t * stride];
  si[t] = t;
  __syncthreads();
  for (int k = 2; k <= 256; k <<= 1) {
    for (int j = k >> 1; j > 0; j >>= 1) {
      int ixj = t ^ j;
      if (ixj > t) {
        bool  up = ((t & k) == 0);
        float a = sv[t], bb = sv[ixj];
        int   ia = si[t], ib = si[ixj];
        bool sw = up ? (a > bb || (a == bb && ia > ib))
                     : (a < bb || (a == bb && ia < ib));
        if (sw) { sv[t] = bb; sv[ixj] = a; si[t] = ib; si[ixj] = ia; }
      }
      __syncthreads();
    }
  }
  X[xbase + (size_t)t * stride]   = sv[t];
  IDX[ibase + (size_t)t * stride] = si[t];
}

// ---------------------------------------------------------------------------
// Pointwise GEMM, activation-read-once: one block per 128-column tile; the
// full weight matrix (Mtot x 48, padded to 64) lives in LDS; the two B
// fragments are hoisted and reused across all Mtot/16 M-tiles.
// ---------------------------------------------------------------------------
__global__ void gemm48_wmma(const float* __restrict__ Wm,
                            const float* __restrict__ bias,
                            const float* __restrict__ X, float* __restrict__ Y,
                            int Mtot) {
  __shared__ __bf16 sA[QKVC_ * 64];   // up to 240x64 bf16 = 30KB
  __shared__ __bf16 sB[128 * 64];     // 16KB
  int nt  = blockIdx.x;
  int b   = blockIdx.y;
  int tid = threadIdx.x;
  for (int s = tid; s < Mtot * 64; s += 256) {
    int m = s >> 6, k = s & 63;
    sA[s] = (__bf16)((k < 48) ? Wm[m * 48 + k] : 0.f);
  }
  const float* Xb = X + (size_t)b * 48 * HW_ + (size_t)nt * 128;
  for (int s = tid; s < 48 * 128; s += 256) {
    int c = s >> 7, j = s & 127;
    sB[j * 64 + c] = (__bf16)Xb[(size_t)c * HW_ + j];
  }
  for (int s = tid; s < 128 * 16; s += 256) {
    int j = s >> 4;
    sB[j * 64 + 48 + (s & 15)] = (__bf16)0.f;
  }
  __syncthreads();
  int lane = tid & 31, wave = tid >> 5;
  v16bf b0 = frag_b_lds(sB, 64, lane, 0, wave * 16);
  v16bf b1 = frag_b_lds(sB, 64, lane, 32, wave * 16);
  int n    = lane & 15;
  int scol = nt * 128 + wave * 16 + n;
  int mb   = (lane >> 4) * 8;
  for (int mt = 0; mt < (Mtot >> 4); ++mt) {
    v8f acc = {};
    v16bf a0 = frag_a_lds(sA + mt * 16 * 64, 64, lane, 0);
    v16bf a1 = frag_a_lds(sA + mt * 16 * 64, 64, lane, 32);
    acc = wmma_bf16(a0, b0, acc);
    acc = wmma_bf16(a1, b1, acc);
#pragma unroll
    for (int r = 0; r < 8; ++r) {
      int o = mt * 16 + mb + r;
      Y[((size_t)b * Mtot + o) * HW_ + scol] = acc[r] + bias[o];
    }
  }
}

// ---------------------------------------------------------------------------
// Depthwise 3x3 'SAME'
// ---------------------------------------------------------------------------
__global__ void dwconv3x3(const float* __restrict__ X, const float* __restrict__ Wd,
                          const float* __restrict__ bd, float* __restrict__ Y) {
  size_t t = (size_t)blockIdx.x * 256 + threadIdx.x;
  int s  = (int)(t & 65535);
  int oc = (int)((t >> 16) % QKVC_);
  int b  = (int)(t / ((size_t)QKVC_ * HW_));
  int y = s >> 8, x = s & 255;
  const float* Xc = X + ((size_t)b * QKVC_ + oc) * HW_;
  const float* w  = Wd + oc * 9;
  float acc = bd[oc];
#pragma unroll
  for (int dy = -1; dy <= 1; ++dy)
#pragma unroll
    for (int dx = -1; dx <= 1; ++dx) {
      int yy = y + dy, xx = x + dx;
      if (yy >= 0 && yy < 256 && xx >= 0 && xx < 256)
        acc += w[(dy + 1) * 3 + (dx + 1)] * Xc[yy * 256 + xx];
    }
  Y[t] = acc;
}

// ---------------------------------------------------------------------------
// v = v * I_cond; init payload
// ---------------------------------------------------------------------------
__global__ void vmul_init(const float* __restrict__ QKV2,
                          const float* __restrict__ Ic, float* __restrict__ VS,
                          int* __restrict__ VI) {
  size_t t = (size_t)blockIdx.x * 256 + threadIdx.x;
  int s = (int)(t & 65535);
  int c = (int)((t >> 16) % 48);
  int b = (int)(t / ((size_t)48 * HW_));
  VS[t] = QKV2[((size_t)b * QKVC_ + 192 + c) * HW_ + s] * Ic[(size_t)b * HW_ + s];
  VI[t] = s;
}

// ---------------------------------------------------------------------------
// LDS-assisted bitonic: a 4096-chunk runs stages k0..k1 (first stage starting
// at sub-pass j0) entirely in LDS. Direction uses the global row position.
// ---------------------------------------------------------------------------
__global__ void bitonic_local(float* __restrict__ V, int* __restrict__ I,
                              int k0, int k1, int j0) {
  __shared__ float sv[4096];
  __shared__ int   si[4096];
  size_t base = (size_t)blockIdx.x * 4096;
  int gbase = (int)(base & 65535);
  int t = threadIdx.x;  // 512
  for (int q = t; q < 4096; q += 512) { sv[q] = V[base + q]; si[q] = I[base + q]; }
  __syncthreads();
  for (int k = k0; k <= k1; k <<= 1) {
    for (int j = (k == k0 ? j0 : (k >> 1)); j >= 1; j >>= 1) {
      for (int p = t; p < 2048; p += 512) {
        int i   = ((p & ~(j - 1)) << 1) | (p & (j - 1));
        int ixj = i | j;
        bool up = (((gbase + i) & k) == 0);
        float a = sv[i], b = sv[ixj];
        int   ia = si[i], ib = si[ixj];
        bool sw = up ? (a > b || (a == b && ia > ib))
                     : (a < b || (a == b && ia < ib));
        if (sw) { sv[i] = b; sv[ixj] = a; si[i] = ib; si[ixj] = ia; }
      }
      __syncthreads();
    }
  }
  for (int q = t; q < 4096; q += 512) { V[base + q] = sv[q]; I[base + q] = si[q]; }
}

// Global pass for j >= 4096 (partners cross 4096-chunks)
__global__ void bitonic_pass(float* __restrict__ V, int* __restrict__ I, int k,
                             int j) {
  size_t t  = (size_t)blockIdx.x * 256 + threadIdx.x;
  int    tt = (int)(t & 32767);
  size_t row = t >> 15;
  int i   = ((tt & ~(j - 1)) << 1) | (tt & (j - 1));
  int ixj = i | j;
  size_t base = row << 16;
  float a = V[base + i], b2 = V[base + ixj];
  int   ia = I[base + i], ib = I[base + ixj];
  bool up = ((i & k) == 0);
  bool sw = up ? (a > b2 || (a == b2 && ia > ib))
               : (a < b2 || (a == b2 && ia < ib));
  if (sw) {
    V[base + i] = b2; V[base + ixj] = a;
    I[base + i] = ib; I[base + ixj] = ia;
  }
}

// ---------------------------------------------------------------------------
// Gather q1,k1,q2,k2 through the sort permutation
// ---------------------------------------------------------------------------
__global__ void gather4(const float* __restrict__ QKV2, const int* __restrict__ VI,
                        float* __restrict__ G) {
  size_t t = (size_t)blockIdx.x * 256 + threadIdx.x;
  int c = (int)((t >> 16) % 48);
  int b = (int)(t / ((size_t)48 * HW_));
  int src = VI[t];
  size_t bbase = (size_t)b * QKVC_ * HW_;
  size_t arrsz = (size_t)B_ * 48 * HW_;
#pragma unroll
  for (int a = 0; a < 4; ++a)
    G[a * arrsz + t] = QKV2[bbase + (size_t)(a * 48 + c) * HW_ + src];
}

// ---------------------------------------------------------------------------
// to_tar + optional l2norm, bf16 tar layout (b,8,48,8192)
// ---------------------------------------------------------------------------
__global__ void tar_convert(const float* __restrict__ src,
                            uint16_t* __restrict__ dstu, int box, int donorm) {
  __shared__ float red[256];
  __bf16* dst = reinterpret_cast<__bf16*>(dstu);
  int row = blockIdx.x;
  int i  = row % 48;
  int hd = (row / 48) % 8;
  int b  = row / (48 * 8);
  int ch = hd * 6 + (i >> 3);
  int r  = i & 7;
  const float* sp = src + ((size_t)(b * 48 + ch) << 16);
  int t = threadIdx.x;
  float scale = 1.f;
  if (donorm) {
    float ss = 0.f;
    for (int j = t; j < HWF_; j += 256) {
      float v = box ? sp[r * HWF_ + j] : sp[j * 8 + r];
      ss += v * v;
    }
    red[t] = ss;
    __syncthreads();
    for (int o = 128; o > 0; o >>= 1) {
      if (t < o) red[t] += red[t + o];
      __syncthreads();
    }
    scale = 1.f / fmaxf(sqrtf(red[0]), 1e-12f);
  }
  __bf16* dp = dst + (size_t)row * HWF_;
  for (int j = t; j < HWF_; j += 256) {
    float v = box ? sp[r * HWF_ + j] : sp[j * 8 + r];
    dp[j] = (__bf16)(v * scale);
  }
}

// ---------------------------------------------------------------------------
// attn += Qn(48xKpart) @ Kn^T, split-K over blockIdx.y (32 partitions of 256).
// Q/K chunks staged via CDNA5 async global->LDS b128; f32 atomics accumulate.
// ---------------------------------------------------------------------------
__global__ void attn_qk(const uint16_t* __restrict__ Qu,
                        const uint16_t* __restrict__ Ku,
                        float* __restrict__ ATT) {
  __shared__ __bf16 sQ[48 * 64];
  __shared__ __bf16 sK[48 * 64];
  const __bf16* Q = reinterpret_cast<const __bf16*>(Qu);
  const __bf16* K = reinterpret_cast<const __bf16*>(Ku);
  int bh   = blockIdx.x;
  int part = blockIdx.y;
  const __bf16* Qb = Q + (size_t)bh * 48 * HWF_;
  const __bf16* Kb = K + (size_t)bh * 48 * HWF_;
  int t = threadIdx.x;  // 288 threads, 9 waves
  int lane = t & 31, wave = t >> 5;
  int mi = wave / 3, ni = wave % 3;
  uint32_t ldsQ = lds_off32(sQ), ldsK = lds_off32(sK);
  v8f acc = {};
  for (int kc = part * 256; kc < part * 256 + 256; kc += 64) {
    __syncthreads();
    for (int id = t; id < 384; id += 288) {
      int rr = id >> 3;
      int kk = (id & 7) * 8;
      async_g2l_b128(ldsQ + (uint32_t)(id << 4), Qb + (size_t)rr * HWF_ + kc + kk);
      async_g2l_b128(ldsK + (uint32_t)(id << 4), Kb + (size_t)rr * HWF_ + kc + kk);
    }
    wait_async0();
    __syncthreads();
#pragma unroll
    for (int ks = 0; ks < 2; ++ks) {
      v16bf a  = frag_a_lds(sQ + mi * 16 * 64, 64, lane, ks * 32);
      v16bf bf = frag_b_lds(sK, 64, lane, ks * 32, ni * 16);
      acc = wmma_bf16(a, bf, acc);
    }
  }
  int n = lane & 15, mb = (lane >> 4) * 8;
#pragma unroll
  for (int r = 0; r < 8; ++r)
    atomicAdd(&ATT[(size_t)bh * 2304 + (mi * 16 + mb + r) * 48 + ni * 16 + n],
              acc[r]);
}

// ---------------------------------------------------------------------------
// softmax_1 with temperature, bf16 convert, K padded 48->64
// ---------------------------------------------------------------------------
__global__ void softmax1_bf16(const float* __restrict__ ATT,
                              const float* __restrict__ temp,
                              uint16_t* __restrict__ AEu) {
  __bf16* AE = reinterpret_cast<__bf16*>(AEu);
  int bh = blockIdx.x;
  float tp = temp[bh & 7];
  int r  = threadIdx.x;
  if (r < 48) {
    const float* row = ATT + (size_t)bh * 2304 + r * 48;
    float e[48];
    float s = 0.f;
    for (int j = 0; j < 48; ++j) { e[j] = __expf(row[j] * tp); s += e[j]; }
    float inv = 1.f / (s + 1.f);
    __bf16* orow = AE + (size_t)bh * 48 * 64 + r * 64;
    for (int j = 0; j < 48; ++j) orow[j] = (__bf16)(e[j] * inv);
    for (int j = 48; j < 64; ++j) orow[j] = (__bf16)0.f;
  }
}

// ---------------------------------------------------------------------------
// O (48x8192) = AE (48x48 pad64) @ Vt (48x8192); N tiled by 128.
// B fragments hoisted out of the M loop.
// ---------------------------------------------------------------------------
__global__ void attn_v(const uint16_t* __restrict__ AEu,
                       const uint16_t* __restrict__ Vtu, float* __restrict__ Ot) {
  __shared__ __bf16 sA[48 * 64];
  __shared__ __bf16 sV[128 * 64];
  const __bf16* AE = reinterpret_cast<const __bf16*>(AEu);
  const __bf16* Vt = reinterpret_cast<const __bf16*>(Vtu);
  int nt = blockIdx.x;
  int bh = blockIdx.y;
  int t  = threadIdx.x;
  const __bf16* Ab = AE + (size_t)bh * 48 * 64;
  const __bf16* Vb = Vt + (size_t)bh * 48 * HWF_ + (size_t)nt * 128;
  uint32_t ldsA = lds_off32(sA);
  for (int id = t; id < 384; id += 256)
    async_g2l_b128(ldsA + (uint32_t)(id << 4), Ab + id * 8);
  for (int s = t; s < 48 * 128; s += 256) {
    int kk = s >> 7, j = s & 127;
    sV[j * 64 + kk] = Vb[(size_t)kk * HWF_ + j];
  }
  for (int s = t; s < 128 * 16; s += 256) {
    int j = s >> 4;
    sV[j * 64 + 48 + (s & 15)] = (__bf16)0.f;
  }
  wait_async0();
  __syncthreads();
  int lane = t & 31, wave = t >> 5;
  int n = lane & 15, mb = (lane >> 4) * 8;
  v16bf b0 = frag_b_lds(sV, 64, lane, 0, wave * 16);
  v16bf b1 = frag_b_lds(sV, 64, lane, 32, wave * 16);
  int col = nt * 128 + wave * 16 + n;
#pragma unroll
  for (int mi = 0; mi < 3; ++mi) {
    v8f acc = {};
    v16bf a0 = frag_a_lds(sA + mi * 16 * 64, 64, lane, 0);
    v16bf a1 = frag_a_lds(sA + mi * 16 * 64, 64, lane, 32);
    acc = wmma_bf16(a0, b0, acc);
    acc = wmma_bf16(a1, b1, acc);
#pragma unroll
    for (int r = 0; r < 8; ++r)
      Ot[(size_t)bh * 48 * HWF_ + (size_t)(mi * 16 + mb + r) * HWF_ + col] =
          acc[r];
  }
}

// ---------------------------------------------------------------------------
// from_tar(O1)*from_tar(O2), scattered back through the sort permutation
// ---------------------------------------------------------------------------
__global__ void prod_scatter(const float* __restrict__ O1,
                             const float* __restrict__ O2,
                             const int* __restrict__ VI, float* __restrict__ P) {
  size_t t = (size_t)blockIdx.x * 256 + threadIdx.x;
  int s = (int)(t & 65535);
  int c = (int)((t >> 16) % 48);
  int b = (int)(t / ((size_t)48 * HW_));
  int hd = c / 6, ci = c % 6;
  int r1 = s >> 13, j1 = s & 8191;
  float o1 = O1[((size_t)(b * 8 + hd) * 48 + ci * 8 + r1) * HWF_ + j1];
  int r2 = s & 7, j2 = s >> 3;
  float o2 = O2[((size_t)(b * 8 + hd) * 48 + ci * 8 + r2) * HWF_ + j2];
  int dst = VI[t];
  P[(((size_t)(b * 48 + c)) << 16) + dst] = o1 * o2;
}

// ---------------------------------------------------------------------------
// Final assembly
// ---------------------------------------------------------------------------
__global__ void final_upper(const float* __restrict__ PJ, float* __restrict__ OUT) {
  size_t t = (size_t)blockIdx.x * 256 + threadIdx.x;
  int s = (int)(t & 65535);
  int c = (int)((t >> 16) % 24);
  int b = (int)(t / ((size_t)24 * HW_));
  size_t idx = ((size_t)(b * 48 + 24 + c) << 16) + s;
  OUT[idx] = PJ[idx];
}

__global__ void final_lower(const float* __restrict__ PJ,
                            const int* __restrict__ IW,
                            const int* __restrict__ IH, float* __restrict__ OUT) {
  size_t t = (size_t)blockIdx.x * 256 + threadIdx.x;
  int j  = (int)(t & 255);
  int hh = (int)((t >> 8) & 255);
  int c  = (int)((t >> 16) % 24);
  int b  = (int)(t / ((size_t)24 * HW_));
  size_t pbase = ((size_t)(b * 48 + c) << 16);
  size_t ibase = ((size_t)(b * 24 + c) << 16);
  float p  = PJ[pbase + hh * 256 + j];
  int wd   = IW[ibase + hh * 256 + j];
  int hdst = IH[ibase + hh * 256 + wd];
  OUT[pbase + hdst * 256 + wd] = p;
}

// ---------------------------------------------------------------------------
extern "C" void kernel_launch(void* const* d_in, const int* in_sizes, int n_in,
                              void* d_out, int out_size, void* d_ws,
                              size_t ws_size, hipStream_t stream) {
  const float* x      = (const float*)d_in[0];
  const float* I_cond = (const float*)d_in[1];
  const float* qkv_w  = (const float*)d_in[2];
  const float* qkv_b  = (const float*)d_in[3];
  const float* dw_w   = (const float*)d_in[4];
  const float* dw_b   = (const float*)d_in[5];
  const float* proj_w = (const float*)d_in[6];
  const float* proj_b = (const float*)d_in[7];
  const float* temp   = (const float*)d_in[8];
  float* out = (float*)d_out;

  const size_t BCHW  = (size_t)B_ * 48 * HW_;
  const size_t BQHW  = (size_t)B_ * QKVC_ * HW_;
  const size_t BHALF = (size_t)B_ * 24 * HW_;
  const size_t TARN  = (size_t)B_ * 8 * 48 * HWF_;

  size_t off = 0;
  auto alloc = [&](size_t bytes) -> char* {
    char* p = (char*)d_ws + off;
    off += (bytes + 255) & ~(size_t)255;
    return p;
  };
  float* xs    = (float*)alloc(BCHW * 4);
  int*   idx_h = (int*)alloc(BHALF * 4);
  int*   idx_w = (int*)alloc(BHALF * 4);
  float* qkv   = (float*)alloc(BQHW * 4);
  float* qkv2  = (float*)alloc(BQHW * 4);
  float* vs    = (float*)alloc(BCHW * 4);
  int*   vi    = (int*)alloc(BCHW * 4);
  uint16_t* qn1 = (uint16_t*)alloc(TARN * 2);
  uint16_t* kn1 = (uint16_t*)alloc(TARN * 2);
  uint16_t* qn2 = (uint16_t*)alloc(TARN * 2);
  uint16_t* kn2 = (uint16_t*)alloc(TARN * 2);
  uint16_t* vt1 = (uint16_t*)alloc(TARN * 2);
  uint16_t* vt2 = (uint16_t*)alloc(TARN * 2);
  float* att1 = (float*)alloc((size_t)B_ * 8 * 2304 * 4);
  float* att2 = (float*)alloc((size_t)B_ * 8 * 2304 * 4);
  uint16_t* ae1 = (uint16_t*)alloc((size_t)B_ * 8 * 48 * 64 * 2);
  uint16_t* ae2 = (uint16_t*)alloc((size_t)B_ * 8 * 48 * 64 * 2);
  if (off > ws_size) return;

  float* gat  = qkv;
  float* o1   = qkv2;
  float* o2   = qkv2 + BCHW;
  float* pj   = qkv2 + 2 * BCHW;
  float* prod = xs;

  // 1) copy x, sort first-half channels along h then w
  hipMemcpyAsync(xs, x, BCHW * 4, hipMemcpyDeviceToDevice, stream);
  sort256_kernel<<<B_ * 24 * 256, 256, 0, stream>>>(xs, idx_h, 0);
  sort256_kernel<<<B_ * 24 * 256, 256, 0, stream>>>(xs, idx_w, 1);

  // 2) qkv 1x1 conv (WMMA, activations read once)
  gemm48_wmma<<<dim3(HW_ / 128, B_), 256, 0, stream>>>(qkv_w, qkv_b, xs, qkv,
                                                       QKVC_);

  // 3) depthwise 3x3
  dwconv3x3<<<(unsigned)(BQHW / 256), 256, 0, stream>>>(qkv, dw_w, dw_b, qkv2);

  // 4) v * I_cond, LDS-assisted bitonic argsort over 65536 (15 launches)
  vmul_init<<<(unsigned)(BCHW / 256), 256, 0, stream>>>(qkv2, I_cond, vs, vi);
  const unsigned nchunks = (unsigned)(BCHW / 4096);
  bitonic_local<<<nchunks, 512, 0, stream>>>(vs, vi, 2, 4096, 1);
  for (int k = 8192; k <= 65536; k <<= 1) {
    for (int j = k >> 1; j >= 4096; j >>= 1)
      bitonic_pass<<<(unsigned)(BCHW / 2 / 256), 256, 0, stream>>>(vs, vi, k, j);
    bitonic_local<<<nchunks, 512, 0, stream>>>(vs, vi, k, k, 2048);
  }

  // 5) gather q1,k1,q2,k2
  gather4<<<(unsigned)(BCHW / 256), 256, 0, stream>>>(qkv2, vi, gat);

  // 6) tar layouts + l2norm + bf16 conversion
  tar_convert<<<B_ * 8 * 48, 256, 0, stream>>>(gat + 0 * BCHW, qn1, 1, 1);
  tar_convert<<<B_ * 8 * 48, 256, 0, stream>>>(gat + 1 * BCHW, kn1, 1, 1);
  tar_convert<<<B_ * 8 * 48, 256, 0, stream>>>(gat + 2 * BCHW, qn2, 0, 1);
  tar_convert<<<B_ * 8 * 48, 256, 0, stream>>>(gat + 3 * BCHW, kn2, 0, 1);
  tar_convert<<<B_ * 8 * 48, 256, 0, stream>>>(vs, vt1, 1, 0);
  tar_convert<<<B_ * 8 * 48, 256, 0, stream>>>(vs, vt2, 0, 0);

  // 7) attention: split-K QK^T (WMMA + async LDS + f32 atomics), softmax+1, @V
  hipMemsetAsync(att1, 0, (size_t)B_ * 8 * 2304 * 4, stream);
  hipMemsetAsync(att2, 0, (size_t)B_ * 8 * 2304 * 4, stream);
  attn_qk<<<dim3(B_ * 8, 32), 288, 0, stream>>>(qn1, kn1, att1);
  attn_qk<<<dim3(B_ * 8, 32), 288, 0, stream>>>(qn2, kn2, att2);
  softmax1_bf16<<<B_ * 8, 64, 0, stream>>>(att1, temp, ae1);
  softmax1_bf16<<<B_ * 8, 64, 0, stream>>>(att2, temp, ae2);
  attn_v<<<dim3(HWF_ / 128, B_ * 8), 256, 0, stream>>>(ae1, vt1, o1);
  attn_v<<<dim3(HWF_ / 128, B_ * 8), 256, 0, stream>>>(ae2, vt2, o2);

  // 8) from_tar product + inverse permutation scatter
  prod_scatter<<<(unsigned)(BCHW / 256), 256, 0, stream>>>(o1, o2, vi, prod);

  // 9) projection GEMM (WMMA)
  gemm48_wmma<<<dim3(HW_ / 128, B_), 256, 0, stream>>>(proj_w, proj_b, prod, pj,
                                                       48);

  // 10) final assembly
  final_upper<<<(unsigned)(BHALF / 256), 256, 0, stream>>>(pj, out);
  final_lower<<<(unsigned)(BHALF / 256), 256, 0, stream>>>(pj, idx_w, idx_h, out);
}
